// EncoderBlock_27470610825321
// MI455X (gfx1250) — matrix operand order repaired and checked by reading, not compiled, not run on hardware
//
#include <hip/hip_runtime.h>
#include <stdint.h>

// ---------------------------------------------------------------------------
// Types
// ---------------------------------------------------------------------------
typedef __bf16 v16bf __attribute__((ext_vector_type(16)));
typedef float  v8f   __attribute__((ext_vector_type(8)));
typedef unsigned short ushort_t;
typedef unsigned int   uint_t;

#define D_MODEL 1024
#define N_HEADS 16
#define DEPTH   64
#define D_FF    4096
#define BATCH   4
#define SEQ     1024
#define MTOT    (BATCH * SEQ)

__device__ __forceinline__ ushort_t f2bf(float f) {
  uint_t u = __float_as_uint(f);
  u += 0x7FFFu + ((u >> 16) & 1u);   // round-to-nearest-even
  return (ushort_t)(u >> 16);
}

union BFrag {
  v16bf v;
  uint4 q[2];
};

// Async global->LDS 16B copy (CDNA5 direct path, ASYNCcnt).  INST_OFFSET is
// added to BOTH the LDS and the global address per the ISA, so one base pair
// covers 64 B with 4 issues.
#define ASYNC_LD16(ldsoff, gptr, o)                                          \
  asm volatile("global_load_async_to_lds_b128 %0, %1, off offset:" #o        \
               ::"v"(ldsoff), "v"(gptr)                                      \
               : "memory")

#define WAIT_ASYNC(n) asm volatile("s_wait_asynccnt " #n ::: "memory")

__device__ __forceinline__ uint_t lds_off32(const void* p) {
  // generic LDS pointer: low 32 bits are the wave-relative LDS byte offset
  return (uint_t)(size_t)p;
}

// ---------------------------------------------------------------------------
// fp32 -> bf16 elementwise convert
// ---------------------------------------------------------------------------
__global__ void cvt_bf16_kernel(const float* __restrict__ X,
                                ushort_t* __restrict__ Y, int n) {
  int i = blockIdx.x * blockDim.x + threadIdx.x;
  if (i < n) Y[i] = f2bf(X[i]);
}

// ---------------------------------------------------------------------------
// W [K][N] fp32  ->  Wt [N][K] bf16  (tile transpose through LDS)
// ---------------------------------------------------------------------------
__global__ void __launch_bounds__(256)
transpose_to_bf16_kernel(const float* __restrict__ W,
                         ushort_t* __restrict__ Wt, int K, int N) {
  __shared__ float tile[32][33];
  const int n0 = blockIdx.x * 32;
  const int k0 = blockIdx.y * 32;
  const int tx = threadIdx.x & 31;
  const int ty = threadIdx.x >> 5;   // 0..7
#pragma unroll
  for (int i = 0; i < 4; i++)
    tile[ty + 8 * i][tx] = W[(size_t)(k0 + ty + 8 * i) * N + n0 + tx];
  __syncthreads();
#pragma unroll
  for (int i = 0; i < 4; i++)
    Wt[(size_t)(n0 + ty + 8 * i) * K + k0 + tx] = f2bf(tile[tx][ty + 8 * i]);
}

// ---------------------------------------------------------------------------
// GEMM:  C[M,N] = A[M,K](bf16) * Bt[N,K](bf16)^T + bias (+ residual fp32)
// Block tile 128x128, 8 waves, wave tile 64x32 (4x2 wmma accumulators).
// K-step 64, double-buffered LDS filled by async global->LDS copies.
// ---------------------------------------------------------------------------
#define GBM 128
#define GBN 128
#define GBK 64
#define LDW 36   // uints (bf16-pairs) per LDS row: 32 used + pad -> 144 B

__global__ void __launch_bounds__(256)
gemm_bf16_kernel(const ushort_t* __restrict__ A,
                 const ushort_t* __restrict__ Bt,
                 const float* __restrict__ bias,
                 const float* __restrict__ residual,
                 float* __restrict__ Cf,
                 ushort_t* __restrict__ Cb,
                 int M, int N, int K) {
  __shared__ uint_t As[2][GBM * LDW];
  __shared__ uint_t Bs[2][GBN * LDW];

  const int tid  = threadIdx.x;
  const int lane = tid & 31;
  const int half = lane >> 4;
  const int l    = lane & 15;
  const int wave = tid >> 5;
  const int wm   = (wave >> 2) * 64;   // wave M offset (0 / 64)
  const int wn   = (wave & 3) * 32;    // wave N offset (0..96)

  const int n0 = blockIdx.x * GBN;
  const int m0 = blockIdx.y * GBM;

  const v8f vzero = {0.f, 0.f, 0.f, 0.f, 0.f, 0.f, 0.f, 0.f};
  v8f acc[4][2];
#pragma unroll
  for (int i = 0; i < 4; i++)
#pragma unroll
    for (int j = 0; j < 2; j++) acc[i][j] = vzero;

  // copy slot: each thread owns 64 B of one row per operand per tile
  const int row = tid >> 1;
  const int seg = tid & 1;
  const ushort_t* gA0 = A  + (size_t)(m0 + row) * K + seg * 32;
  const ushort_t* gB0 = Bt + (size_t)(n0 + row) * K + seg * 32;
  const int lslot = row * LDW + seg * 16;

  const int NK = K >> 6;   // K / 64

  // prologue: prefetch tile 0 into buffer 0
  {
    uint_t la = lds_off32(&As[0][lslot]);
    uint_t lb = lds_off32(&Bs[0][lslot]);
    ASYNC_LD16(la, gA0, 0);  ASYNC_LD16(la, gA0, 16);
    ASYNC_LD16(la, gA0, 32); ASYNC_LD16(la, gA0, 48);
    ASYNC_LD16(lb, gB0, 0);  ASYNC_LD16(lb, gB0, 16);
    ASYNC_LD16(lb, gB0, 32); ASYNC_LD16(lb, gB0, 48);
  }

  for (int kt = 0; kt < NK; ++kt) {
    const int cur = kt & 1;
    if (kt + 1 < NK) {
      // prefetch next tile into the other buffer (writers were fenced by the
      // trailing barrier of the previous iteration)
      const ushort_t* gA = gA0 + (size_t)(kt + 1) * GBK;
      const ushort_t* gB = gB0 + (size_t)(kt + 1) * GBK;
      uint_t la = lds_off32(&As[cur ^ 1][lslot]);
      uint_t lb = lds_off32(&Bs[cur ^ 1][lslot]);
      ASYNC_LD16(la, gA, 0);  ASYNC_LD16(la, gA, 16);
      ASYNC_LD16(la, gA, 32); ASYNC_LD16(la, gA, 48);
      ASYNC_LD16(lb, gB, 0);  ASYNC_LD16(lb, gB, 16);
      ASYNC_LD16(lb, gB, 32); ASYNC_LD16(lb, gB, 48);
      // async loads complete in order: <=8 outstanding => current tile landed
      WAIT_ASYNC(0x8);
    } else {
      WAIT_ASYNC(0x0);
    }
    __syncthreads();

    const uint_t* Ab = As[cur];
    const uint_t* Bb = Bs[cur];
#pragma unroll
    for (int ks = 0; ks < 2; ++ks) {
      BFrag a[4], b[2];
#pragma unroll
      for (int i = 0; i < 4; i++) {  // A layout: pairs (c&3)+half*4+(c>>2)*8
        const uint_t* p = &Ab[(wm + i * 16 + l) * LDW + ks * 16 + half * 4];
        a[i].q[0] = *(const uint4*)p;
        a[i].q[1] = *(const uint4*)(p + 8);
      }
#pragma unroll
      for (int j = 0; j < 2; j++) {  // B layout: pairs (c&3)+half*4+(c>>2)*8
        const uint_t* p = &Bb[(wn + j * 16 + l) * LDW + ks * 16 + half * 4];
        b[j].q[0] = *(const uint4*)p;
        b[j].q[1] = *(const uint4*)(p + 8);
      }
#pragma unroll
      for (int i = 0; i < 4; i++)
#pragma unroll
        for (int j = 0; j < 2; j++)
          acc[i][j] = __builtin_amdgcn_wmma_f32_16x16x32_bf16(
              false, a[i].v, false, b[j].v, (short)0, acc[i][j], false, false);
    }
    __syncthreads();  // readers done before next prefetch overwrites buffer
  }

  // epilogue: bias + optional residual, fp32 and/or bf16 store
#pragma unroll
  for (int i = 0; i < 4; i++) {
    const int rbase = m0 + wm + i * 16 + half * 8;
#pragma unroll
    for (int j = 0; j < 2; j++) {
      const int col = n0 + wn + j * 16 + l;
      const float bv = bias ? bias[col] : 0.0f;
#pragma unroll
      for (int r = 0; r < 8; r++) {
        size_t idx = (size_t)(rbase + r) * N + col;
        float v = acc[i][j][r] + bv;
        if (residual) v += residual[idx];
        if (Cf) Cf[idx] = v;
        if (Cb) Cb[idx] = f2bf(v);
      }
    }
  }
}

// ---------------------------------------------------------------------------
// Flash attention: one block = (b,h) x 64 q-rows. 4 waves, 16 q-rows/wave.
// S = Q K^T / 8 + mask*-1e9 ; online softmax ; O += P V ; O /= l
// ---------------------------------------------------------------------------
#define ALDP 36   // 32 bf16-pairs used + pad -> 144 B row stride

__global__ void __launch_bounds__(128)
attention_kernel(const ushort_t* __restrict__ Q,
                 const ushort_t* __restrict__ Kmat,
                 const ushort_t* __restrict__ V,
                 const float* __restrict__ mask,
                 ushort_t* __restrict__ Ctx) {
  __shared__ uint_t q_s [64 * ALDP];
  __shared__ uint_t k_s [64 * ALDP];
  __shared__ uint_t vt_s[64 * ALDP];   // V transposed: rows = d, pairs over t
  __shared__ uint_t p_s [64 * ALDP];   // per-wave 16-row P staging

  const int tid  = threadIdx.x;
  const int lane = tid & 31;
  const int half = lane >> 4;
  const int l    = lane & 15;
  const int wave = tid >> 5;

  const int bh = blockIdx.y;
  const int b  = bh >> 4;
  const int h  = bh & 15;
  const int q0 = blockIdx.x * 64;

  const ushort_t* Qg = Q    + ((size_t)b * SEQ) * D_MODEL + h * DEPTH;
  const ushort_t* Kg = Kmat + ((size_t)b * SEQ) * D_MODEL + h * DEPTH;
  const ushort_t* Vg = V    + ((size_t)b * SEQ) * D_MODEL + h * DEPTH;
  const float*    mg = mask + (size_t)b * SEQ;

  const int row = tid >> 1, seg = tid & 1;   // copy slot: 64 B per thread

  // load Q tile: 64 rows x 64 bf16
  {
    const uint4* src = (const uint4*)(Qg + (size_t)(q0 + row) * D_MODEL);
    uint4* dst = (uint4*)&q_s[row * ALDP + seg * 16];
#pragma unroll
    for (int c = 0; c < 4; c++) dst[c] = src[seg * 4 + c];
  }
  __syncthreads();

  BFrag qf[2];
#pragma unroll
  for (int kk = 0; kk < 2; kk++) {   // A-layout fragments of Q (K = 64)
    const uint_t* p = &q_s[(wave * 16 + l) * ALDP + kk * 16 + half * 4];
    qf[kk].q[0] = *(const uint4*)p;
    qf[kk].q[1] = *(const uint4*)(p + 8);
  }

  const v8f vzero = {0.f, 0.f, 0.f, 0.f, 0.f, 0.f, 0.f, 0.f};
  v8f o_acc[4];
  float m_run[8], l_run[8];
#pragma unroll
  for (int j = 0; j < 4; j++) o_acc[j] = vzero;
#pragma unroll
  for (int r = 0; r < 8; r++) { m_run[r] = -3.0e38f; l_run[r] = 0.f; }

  for (int t0 = 0; t0 < SEQ; t0 += 64) {
    __syncthreads();  // protect k_s / vt_s from previous iteration readers
    {  // K tile via async global->LDS: rows = t, pairs over d
      const ushort_t* src = Kg + (size_t)(t0 + row) * D_MODEL + seg * 32;
      uint_t la = lds_off32(&k_s[row * ALDP + seg * 16]);
      ASYNC_LD16(la, src, 0);  ASYNC_LD16(la, src, 16);
      ASYNC_LD16(la, src, 32); ASYNC_LD16(la, src, 48);
    }
    {  // V tile transposed: element (d, t) at ushort offset d*72 + t
      ushort_t* vt = (ushort_t*)vt_s;
#pragma unroll
      for (int it = 0; it < 16; it++) {
        const int id   = tid * 16 + it;   // 0..2047
        const int trow = id >> 5;         // t in tile
        const int dp   = id & 31;         // d-pair
        uint_t pv = ((const uint_t*)(Vg + (size_t)(t0 + trow) * D_MODEL))[dp];
        vt[(2 * dp    ) * (ALDP * 2) + trow] = (ushort_t)(pv & 0xFFFFu);
        vt[(2 * dp + 1) * (ALDP * 2) + trow] = (ushort_t)(pv >> 16);
      }
    }
    WAIT_ASYNC(0x0);
    __syncthreads();

    // S = Q K^T  (wave: 16 x 64)
    float sreg[4][8];
#pragma unroll
    for (int j = 0; j < 4; j++) {
      v8f s = vzero;
#pragma unroll
      for (int kk = 0; kk < 2; kk++) {
        BFrag kb;  // B-layout: rows = t-col, pairs over d
        const uint_t* p = &k_s[(j * 16 + l) * ALDP + kk * 16 + half * 4];
        kb.q[0] = *(const uint4*)p;
        kb.q[1] = *(const uint4*)(p + 8);
        s = __builtin_amdgcn_wmma_f32_16x16x32_bf16(
            false, qf[kk].v, false, kb.v, (short)0, s, false, false);
      }
      const float mv = mg[t0 + j * 16 + l] * -1.0e9f;
#pragma unroll
      for (int r = 0; r < 8; r++) sreg[j][r] = s[r] * 0.125f + mv;
    }

    // online softmax per row (row = half*8 + r); reductions stay in 16-lane half
#pragma unroll
    for (int r = 0; r < 8; r++) {
      float cur = fmaxf(fmaxf(sreg[0][r], sreg[1][r]),
                        fmaxf(sreg[2][r], sreg[3][r]));
#pragma unroll
      for (int m = 1; m < 16; m <<= 1) cur = fmaxf(cur, __shfl_xor(cur, m, 32));
      const float mnew = fmaxf(m_run[r], cur);
      const float corr = __expf(m_run[r] - mnew);
      float rsum = 0.f;
#pragma unroll
      for (int j = 0; j < 4; j++) {
        float pj = __expf(sreg[j][r] - mnew);
        sreg[j][r] = pj;
        rsum += pj;
      }
#pragma unroll
      for (int m = 1; m < 16; m <<= 1) rsum += __shfl_xor(rsum, m, 32);
      l_run[r] = l_run[r] * corr + rsum;
      m_run[r] = mnew;
#pragma unroll
      for (int j2 = 0; j2 < 4; j2++) o_acc[j2][r] *= corr;
    }

    // stage P (bf16) in per-wave LDS region, A-layout rows
    {
      ushort_t* pw = (ushort_t*)&p_s[wave * 16 * ALDP];
#pragma unroll
      for (int j = 0; j < 4; j++)
#pragma unroll
        for (int r = 0; r < 8; r++)
          pw[(half * 8 + r) * (ALDP * 2) + j * 16 + l] = f2bf(sreg[j][r]);
    }
    __syncthreads();

    // O += P V
#pragma unroll
    for (int kk = 0; kk < 2; kk++) {
      BFrag pf;  // A-layout fragment of P (K = t)
      const uint_t* pp = &p_s[(wave * 16 + l) * ALDP + kk * 16 + half * 4];
      pf.q[0] = *(const uint4*)pp;
      pf.q[1] = *(const uint4*)(pp + 8);
#pragma unroll
      for (int j2 = 0; j2 < 4; j2++) {
        BFrag vf;  // B-layout: rows = d-col, pairs over t
        const uint_t* vp = &vt_s[(j2 * 16 + l) * ALDP + kk * 16 + half * 4];
        vf.q[0] = *(const uint4*)vp;
        vf.q[1] = *(const uint4*)(vp + 8);
        o_acc[j2] = __builtin_amdgcn_wmma_f32_16x16x32_bf16(
            false, pf.v, false, vf.v, (short)0, o_acc[j2], false, false);
      }
    }
  }

  // finalize: divide by running sum, store bf16 ctx in [b*S+s][h*64+d]
#pragma unroll
  for (int j2 = 0; j2 < 4; j2++)
#pragma unroll
    for (int r = 0; r < 8; r++) {
      const float o = o_acc[j2][r] / l_run[r];
      const size_t orow = (size_t)b * SEQ + q0 + wave * 16 + half * 8 + r;
      Ctx[orow * D_MODEL + h * DEPTH + j2 * 16 + l] = f2bf(o);
    }
}

// ---------------------------------------------------------------------------
// Row LayerNorm over D_MODEL; optional fp32 and bf16 outputs (in-place safe)
// ---------------------------------------------------------------------------
__global__ void __launch_bounds__(256)
layernorm_kernel(const float* __restrict__ X,
                 const float* __restrict__ gamma,
                 const float* __restrict__ beta,
                 float* __restrict__ Yf,
                 ushort_t* __restrict__ Yb) {
  __shared__ float red[16];
  const int row = blockIdx.x;
  const int tid = threadIdx.x;
  const float* xr = X + (size_t)row * D_MODEL;
  float x[4];
  float s = 0.f, sq = 0.f;
#pragma unroll
  for (int i = 0; i < 4; i++) {
    x[i] = xr[tid + 256 * i];
    s += x[i];
    sq += x[i] * x[i];
  }
#pragma unroll
  for (int m = 1; m < 32; m <<= 1) {
    s  += __shfl_xor(s, m, 32);
    sq += __shfl_xor(sq, m, 32);
  }
  const int wave = tid >> 5, lane = tid & 31;
  if (lane == 0) { red[wave] = s; red[8 + wave] = sq; }
  __syncthreads();
  if (tid == 0) {
    float ts = 0.f, tq = 0.f;
#pragma unroll
    for (int w = 0; w < 8; w++) { ts += red[w]; tq += red[8 + w]; }
    red[0] = ts; red[8] = tq;
  }
  __syncthreads();
  const float mean = red[0] * (1.0f / D_MODEL);
  const float var  = red[8] * (1.0f / D_MODEL) - mean * mean;
  const float rstd = rsqrtf(var + 1e-6f);
#pragma unroll
  for (int i = 0; i < 4; i++) {
    const int c = tid + 256 * i;
    const float y = (x[i] - mean) * rstd * gamma[c] + beta[c];
    if (Yf) Yf[(size_t)row * D_MODEL + c] = y;
    if (Yb) Yb[(size_t)row * D_MODEL + c] = f2bf(y);
  }
}

// ---------------------------------------------------------------------------
// Host orchestration
// ---------------------------------------------------------------------------
extern "C" void kernel_launch(void* const* d_in, const int* in_sizes, int n_in,
                              void* d_out, int out_size, void* d_ws,
                              size_t ws_size, hipStream_t stream) {
  (void)in_sizes; (void)n_in; (void)out_size; (void)ws_size;
  const float* x    = (const float*)d_in[0];
  const float* mask = (const float*)d_in[1];
  const float* wq = (const float*)d_in[2];   const float* bq  = (const float*)d_in[3];
  const float* wk = (const float*)d_in[4];   const float* bk  = (const float*)d_in[5];
  const float* wv = (const float*)d_in[6];   const float* bv  = (const float*)d_in[7];
  const float* wo = (const float*)d_in[8];   const float* bo  = (const float*)d_in[9];
  const float* w1 = (const float*)d_in[10];  const float* b1  = (const float*)d_in[11];
  const float* w2 = (const float*)d_in[12];  const float* b2  = (const float*)d_in[13];
  const float* g1 = (const float*)d_in[14];  const float* be1 = (const float*)d_in[15];
  const float* g2 = (const float*)d_in[16];  const float* be2 = (const float*)d_in[17];
  float* out = (float*)d_out;

  char* ws = (char*)d_ws;
  const size_t MB = 1024u * 1024u;
  ushort_t* Xb  = (ushort_t*)(ws + 0 * MB);    // 8 MB
  ushort_t* Wqt = (ushort_t*)(ws + 8 * MB);    // 2 MB each
  ushort_t* Wkt = (ushort_t*)(ws + 10 * MB);
  ushort_t* Wvt = (ushort_t*)(ws + 12 * MB);
  ushort_t* Wot = (ushort_t*)(ws + 14 * MB);
  ushort_t* W1t = (ushort_t*)(ws + 16 * MB);   // 8 MB
  ushort_t* W2t = (ushort_t*)(ws + 24 * MB);   // 8 MB
  ushort_t* Qb  = (ushort_t*)(ws + 32 * MB);   // 8 MB
  ushort_t* Kb  = (ushort_t*)(ws + 40 * MB);   // 8 MB
  ushort_t* Vb  = (ushort_t*)(ws + 48 * MB);   // 8 MB
  ushort_t* Ctx = (ushort_t*)(ws + 56 * MB);   // 8 MB
  float*    T1  = (float*)   (ws + 64 * MB);   // 16 MB fp32 (z+res, then out1)
  ushort_t* O1b = (ushort_t*)(ws + 80 * MB);   // 8 MB
  ushort_t* FF1 = (ushort_t*)(ws + 32 * MB);   // 32 MB, reuses dead Q/K/V/Ctx
  float*    T2  = (float*)   (ws + 88 * MB);   // 16 MB fp32

  // 1) convert inputs to bf16
  cvt_bf16_kernel<<<(MTOT * D_MODEL + 255) / 256, 256, 0, stream>>>(
      x, Xb, MTOT * D_MODEL);

  // 2) transpose + convert weights: W[K][N] -> Wt[N][K] bf16
  transpose_to_bf16_kernel<<<dim3(D_MODEL / 32, D_MODEL / 32), 256, 0, stream>>>(
      wq, Wqt, D_MODEL, D_MODEL);
  transpose_to_bf16_kernel<<<dim3(D_MODEL / 32, D_MODEL / 32), 256, 0, stream>>>(
      wk, Wkt, D_MODEL, D_MODEL);
  transpose_to_bf16_kernel<<<dim3(D_MODEL / 32, D_MODEL / 32), 256, 0, stream>>>(
      wv, Wvt, D_MODEL, D_MODEL);
  transpose_to_bf16_kernel<<<dim3(D_MODEL / 32, D_MODEL / 32), 256, 0, stream>>>(
      wo, Wot, D_MODEL, D_MODEL);
  transpose_to_bf16_kernel<<<dim3(D_FF / 32, D_MODEL / 32), 256, 0, stream>>>(
      w1, W1t, D_MODEL, D_FF);
  transpose_to_bf16_kernel<<<dim3(D_MODEL / 32, D_FF / 32), 256, 0, stream>>>(
      w2, W2t, D_FF, D_MODEL);

  // 3) QKV projections (bf16 out)
  dim3 gqkv(D_MODEL / GBN, MTOT / GBM);
  gemm_bf16_kernel<<<gqkv, 256, 0, stream>>>(Xb, Wqt, bq, nullptr, nullptr, Qb,
                                             MTOT, D_MODEL, D_MODEL);
  gemm_bf16_kernel<<<gqkv, 256, 0, stream>>>(Xb, Wkt, bk, nullptr, nullptr, Kb,
                                             MTOT, D_MODEL, D_MODEL);
  gemm_bf16_kernel<<<gqkv, 256, 0, stream>>>(Xb, Wvt, bv, nullptr, nullptr, Vb,
                                             MTOT, D_MODEL, D_MODEL);

  // 4) fused flash attention
  attention_kernel<<<dim3(SEQ / 64, BATCH * N_HEADS), 128, 0, stream>>>(
      Qb, Kb, Vb, mask, Ctx);

  // 5) output projection + residual(inputs) -> fp32 T1
  gemm_bf16_kernel<<<gqkv, 256, 0, stream>>>(Ctx, Wot, bo, x, T1, nullptr,
                                             MTOT, D_MODEL, D_MODEL);

  // 6) LN1: T1 -> T1 (fp32, in-place) + O1b (bf16)
  layernorm_kernel<<<MTOT, 256, 0, stream>>>(T1, g1, be1, T1, O1b);

  // 7) FFN1 (no activation per reference) -> bf16 FF1
  gemm_bf16_kernel<<<dim3(D_FF / GBN, MTOT / GBM), 256, 0, stream>>>(
      O1b, W1t, b1, nullptr, nullptr, FF1, MTOT, D_FF, D_MODEL);

  // 8) FFN2 + residual(out1) -> fp32 T2
  gemm_bf16_kernel<<<gqkv, 256, 0, stream>>>(FF1, W2t, b2, T1, T2, nullptr,
                                             MTOT, D_MODEL, D_FF);

  // 9) LN2 -> final output
  layernorm_kernel<<<MTOT, 256, 0, stream>>>(T2, g2, be2, out, nullptr);
}
